// Lstm2D_61770219651748
// MI455X (gfx1250) — matrix operand order repaired and checked by reading, not compiled, not run on hardware
//
#include <hip/hip_runtime.h>
#include <math.h>

// ---------------------------------------------------------------------------
// 2D-LSTM over 8x8 patches: B=64, C=3, H=W=256 -> T=1024 steps, NC=512.
// gates = [x_t | h_{t-32} | h_{t-1}] @ [W_ih;W_hh]^T + bias  (K = 192+512+512)
// Persistent single kernel: 32 blocks x 512 threads, device-wide barrier per
// step, bf16 WMMA (v_wmma_f32_16x16x32_bf16).
//  - Weights packed once into CDNA5 B-fragment layout, then each block copies
//    its 152KB B slice into LDS (320KB/WGP) and streams it via ds_load_b128
//    for all 1024 steps.
//  - Recurrent input scatter-packed in bf16 A-fragment layout each step
//    (shared across blocks through L2).
//  - Cell state c lives in registers for all 1024 steps; pointwise phase uses
//    branch-free exp-based sigmoid/tanh (v_exp_f32 + v_rcp_f32).
//  - K-loop trip count is laundered through inline asm so the compiler cannot
//    fully unroll + hoist the (t-invariant) B fragments into spilled VGPRs.
// ---------------------------------------------------------------------------

typedef __attribute__((ext_vector_type(16))) __bf16 v16bf;
typedef __attribute__((ext_vector_type(8)))  float  v8f;

#define BATCH  64
#define NCELLS 512
#define FDIM   192
#define KDIM   1216          // 192 + 512 + 512
#define KT     38            // KDIM / 32
#define KTH    19            // KT / 2
#define TSTEPS 1024
#define NBLK   32            // one block per 16-cell tile
#define NTHR   512           // 16 waves

// dynamic LDS layout (bytes)
#define SB_BYTES (4 * KT * 32 * 16 * 2)     // 155,648 : B fragments (bf16)
#define SG_BYTES (4 * 16 * BATCH * 4)       //  16,384 : i,f,g,o gates (f32)
#define SMEM_BYTES (SB_BYTES + SG_BYTES)    // 172,032 (< 320KB WGP LDS)

// ws layout (bytes)
#define OFF_BPACK 0                         // 128 ntiles * 38 ktiles * 1024B = 4,980,736
#define OFF_APACK 4980736                   // 4 mtiles * 38 ktiles * 1024B  =   155,648
#define OFF_BIAS  (4980736 + 155648)        // 2048 * 4
#define OFF_CTR   (4980736 + 155648 + 8192) // 4

__device__ __forceinline__ unsigned short f2bf(float f) {
  unsigned u = __float_as_uint(f);
  u += 0x7fffu + ((u >> 16) & 1u);          // round-to-nearest-even
  return (unsigned short)(u >> 16);
}

// Branch-free fast transcendentals: __expf -> v_exp_f32, amdgcn_rcpf -> v_rcp_f32.
__device__ __forceinline__ float fast_sigmoid(float x) {
  return __builtin_amdgcn_rcpf(1.f + __expf(-x));
}
__device__ __forceinline__ float fast_tanh(float x) {
  return 1.f - 2.f * __builtin_amdgcn_rcpf(1.f + __expf(2.f * x));
}

// Scatter one element (m, k) of the 64x1216 A matrix into the packed
// 16-bit A-fragment layout (CDNA5 ISA 7.12.2, 16-bit A 16x32):
// lane = half*16 + m%16 ; within-lane half index i recovered from k%32.
__device__ __forceinline__ void scatterA(unsigned short* Ap, int m, int k,
                                         unsigned short v) {
  int mt = m >> 4, ml = m & 15;
  int kt = k >> 5, kl = k & 31;
  int half = (kl >> 3) & 1;
  int i = ((kl >> 4) << 3) | (kl & 7);
  int lane = half * 16 + ml;
  Ap[(((mt * KT) + kt) * 32 + lane) * 16 + i] = v;
}

// ---- one-shot prep: pack weights to bf16 B fragments, bias, reset barrier --
__global__ void prep_kernel(const float* __restrict__ Wih,
                            const float* __restrict__ Whh,
                            const float* __restrict__ bih,
                            const float* __restrict__ bhh,
                            unsigned short* __restrict__ Bp,
                            float* __restrict__ bias,
                            unsigned int* __restrict__ ctr) {
  int tid = blockIdx.x * blockDim.x + threadIdx.x;
  if (tid == 0) *ctr = 0u;
  if (tid < 4 * NCELLS) bias[tid] = bih[tid] + bhh[tid];
  if (tid < 4 * NCELLS * KDIM) {
    int row = tid / KDIM, k = tid - row * KDIM;   // row = gate output index
    float v = (k < 704) ? Wih[row * 704 + k] : Whh[row * NCELLS + (k - 704)];
    // 16-bit B fragment layout: lane = half*16 + n%16, K within half = i
    int nt = row >> 4, nl = row & 15;
    int kt = k >> 5, kl = k & 31;
    int half = (kl >> 3) & 1, i = ((kl >> 4) << 3) | (kl & 7);
    int lane = half * 16 + nl;
    Bp[(((nt * KT) + kt) * 32 + lane) * 16 + i] = f2bf(v);
  }
}

__device__ __forceinline__ void grid_barrier(unsigned int* ctr, unsigned epoch) {
  __syncthreads();
  if (threadIdx.x == 0) {
    __hip_atomic_fetch_add(ctr, 1u, __ATOMIC_RELEASE, __HIP_MEMORY_SCOPE_AGENT);
    while (__hip_atomic_load(ctr, __ATOMIC_ACQUIRE, __HIP_MEMORY_SCOPE_AGENT) <
           (unsigned)NBLK * epoch)
      __builtin_amdgcn_s_sleep(2);
  }
  __syncthreads();
}

// ---- persistent 2D-LSTM kernel --------------------------------------------
__global__ void __launch_bounds__(NTHR, 1)
lstm2d_kernel(const float* __restrict__ x,
              const unsigned short* __restrict__ Bp_,
              unsigned short* __restrict__ Ap_,
              const float* __restrict__ bias,
              unsigned int* __restrict__ ctr,
              float* __restrict__ out) {
  extern __shared__ char smem[];            // [sB: 152KB bf16][sg: 16KB f32]
  char*  sB = smem;
  float* sg = (float*)(smem + SB_BYTES);

  const int blk  = blockIdx.x;              // cell tile: cells [blk*16, blk*16+16)
  const int tid  = threadIdx.x;
  const int wave = tid >> 5, lane = tid & 31;
  const int gq = wave >> 2, mt = wave & 3;  // wave -> (gate quadrant, M tile)
  const v16bf* Ap = (const v16bf*)Ap_;
  float creg[2] = {0.f, 0.f};               // cell state, lives in registers

  // ---- copy this block's B slice (4 contiguous 38KB chunks) into LDS ------
  {
    const int chunk = KT * 512 / 8;         // uint4 per gate chunk = 2432
    for (int idx = tid; idx < 4 * chunk; idx += NTHR) {
      int g = idx / chunk, r = idx - g * chunk;
      ((uint4*)sB)[idx] =
          ((const uint4*)(Bp_ + (size_t)(g * NBLK + blk) * KT * 512))[r];
    }
  }

  // ---- pack A for t=0: x patch (0,0); prev2 = h = 0 -----------------------
  for (int idx = tid; idx < 22 * BATCH; idx += NTHR) {  // cols [blk*22, +22)
    int c2 = idx >> 6, m = idx & 63;
    int k = blk * 22 + c2;                  // 32 blocks * 22 = 704 cols exactly
    float v = 0.f;
    if (k < FDIM) {
      int cch = k >> 6, rem = k & 63, py = rem >> 3, px = rem & 7;
      v = x[((m * 3 + cch) * 256 + py) * 256 + px];     // ty=0, tx=0
    }
    scatterA(Ap_, m, k, f2bf(v));
  }
  for (int q = 0; q < 2; ++q) {             // zero this block's h section
    int e = tid + q * NTHR;
    int m = e & 63, nl = e >> 6;
    scatterA(Ap_, m, 704 + blk * 16 + nl, (unsigned short)0);
  }
  grid_barrier(ctr, 1u);                    // also covers the LDS B fill

#pragma unroll 1
  for (int t = 0; t < TSTEPS; ++t) {
    // -- GEMM phase: wave computes 16x16 tile (gate gq, mtile mt) -----------
    // A streams from L2 (shared by all blocks); B streams from LDS.
    // Two independent accumulators over the two K halves.
    v8f acc0 = {}, acc1 = {};
    const v16bf* afrag = Ap + (mt * KT) * 32 + lane;
    const char*  bbase = sB + ((size_t)(gq * KT) * 32 + lane) * 32;
    int ktc = KTH;
    asm volatile("" : "+s"(ktc));           // opaque trip count: no full unroll
#pragma unroll 1
    for (int kt = 0; kt < ktc; ++kt) {
      v16bf a0 = afrag[kt * 32];
      v16bf a1 = afrag[(kt + KTH) * 32];
      v16bf b0 = *(const v16bf*)(bbase + (size_t)kt * 1024);
      v16bf b1 = *(const v16bf*)(bbase + (size_t)(kt + KTH) * 1024);
      acc0 = __builtin_amdgcn_wmma_f32_16x16x32_bf16(
          false, a0, false, b0, (short)0, acc0, false, false);
      acc1 = __builtin_amdgcn_wmma_f32_16x16x32_bf16(
          false, a1, false, b1, (short)0, acc1, false, false);
    }
    {   // C layout: VGPR r -> M = r + 8*(lane>>4); N = lane&15
      int NL = lane & 15;
      int Mg = mt * 16 + (lane >> 4) * 8;
#pragma unroll
      for (int r = 0; r < 8; ++r)
        sg[(gq * 16 + NL) * BATCH + Mg + r] = acc0[r] + acc1[r];
    }
    __syncthreads();

    // -- cell phase: LSTM pointwise + scatter h (bf16) into next A ----------
    const int tn = t + 1;
#pragma unroll
    for (int q = 0; q < 2; ++q) {
      int e = tid + q * NTHR;
      int m = e & 63, nl = e >> 6;
      int ng = blk * 16 + nl;
      float ig = sg[(0 * 16 + nl) * BATCH + m] + bias[ng];
      float fg = sg[(1 * 16 + nl) * BATCH + m] + bias[NCELLS + ng];
      float gg = sg[(2 * 16 + nl) * BATCH + m] + bias[2 * NCELLS + ng];
      float og = sg[(3 * 16 + nl) * BATCH + m] + bias[3 * NCELLS + ng];
      float cc = fast_sigmoid(fg) * creg[q] + fast_sigmoid(ig) * fast_tanh(gg);
      creg[q] = cc;
      float h = fast_sigmoid(og) * fast_tanh(cc);
      out[(m * NCELLS + ng) * TSTEPS + t] = h;       // (B, NC, sy, sx) layout
      scatterA(Ap_, m, 704 + ng, f2bf(h));           // h_{t} for step t+1
    }

    if (tn < TSTEPS) {
      // -- pack x_{t+1} and prev2 = h_{t-31} (read back from out) -----------
      int ty = tn >> 5, tx = tn & 31;
      for (int idx = tid; idx < 22 * BATCH; idx += NTHR) {
        int c2 = idx >> 6, m = idx & 63;
        int k = blk * 22 + c2;
        float v;
        if (k < FDIM) {
          int cch = k >> 6, rem = k & 63, py = rem >> 3, px = rem & 7;
          v = x[((m * 3 + cch) * 256 + (ty * 8 + py)) * 256 + (tx * 8 + px)];
        } else {
          v = (tn >= 32) ? out[(m * NCELLS + (k - FDIM)) * TSTEPS + (tn - 32)]
                         : 0.f;
        }
        scatterA(Ap_, m, k, f2bf(v));
      }
      grid_barrier(ctr, (unsigned)(t + 2));   // A_pack visible device-wide
    }
  }
}

// ---------------------------------------------------------------------------
extern "C" void kernel_launch(void* const* d_in, const int* in_sizes, int n_in,
                              void* d_out, int out_size, void* d_ws,
                              size_t ws_size, hipStream_t stream) {
  (void)in_sizes; (void)n_in; (void)out_size; (void)ws_size;
  const float* x   = (const float*)d_in[0];
  const float* Wih = (const float*)d_in[1];
  const float* Whh = (const float*)d_in[2];
  const float* bih = (const float*)d_in[3];
  const float* bhh = (const float*)d_in[4];

  char* ws = (char*)d_ws;
  unsigned short* Bp  = (unsigned short*)(ws + OFF_BPACK);
  unsigned short* Ap  = (unsigned short*)(ws + OFF_APACK);
  float*         bias = (float*)(ws + OFF_BIAS);
  unsigned int*  ctr  = (unsigned int*)(ws + OFF_CTR);

  int n = 4 * NCELLS * KDIM;
  prep_kernel<<<(n + 255) / 256, 256, 0, stream>>>(Wih, Whh, bih, bhh, Bp, bias,
                                                   ctr);
  lstm2d_kernel<<<NBLK, NTHR, SMEM_BYTES, stream>>>(x, Bp, Ap, bias, ctr,
                                                    (float*)d_out);
}
